// LRSSM_71579924955200
// MI455X (gfx1250) — matrix operand order repaired and checked by reading, not compiled, not run on hardware
//
#include <hip/hip_runtime.h>

#define BATCH  16
#define HDIM   256
#define NSTATE 32
#define LSEQ   4096
#define C2     512              // 2H
#define HN     (HDIM * NSTATE)  // 8192
#define CHUNK  64
#define NCHUNK (LSEQ / CHUNK)   // 64
#define NCOL   (BATCH * NCHUNK) // 1024 columns per head for the chunk GEMMs

typedef __attribute__((ext_vector_type(16))) __bf16 v16bf;
typedef __attribute__((ext_vector_type(2)))  __bf16 v2bf;
typedef __attribute__((ext_vector_type(8)))  float  v8f;

__device__ inline unsigned short f2bf(float x) {
    __bf16 b = (__bf16)x;
    return __builtin_bit_cast(unsigned short, b);
}
__device__ inline float bf2f(unsigned short s) {
    return (float)__builtin_bit_cast(__bf16, s);
}
// pack two floats into one dword of bf16 (lowers to v_cvt_pk_bf16_f32)
__device__ inline unsigned pk2bf(float lo, float hi) {
    v2bf t; t[0] = (__bf16)lo; t[1] = (__bf16)hi;
    return __builtin_bit_cast(unsigned, t);
}
__device__ inline void store8bf(unsigned short* dst, v8f acc) {
    uint4 v;
    v.x = pk2bf(acc[0], acc[1]);
    v.y = pk2bf(acc[2], acc[3]);
    v.z = pk2bf(acc[4], acc[5]);
    v.w = pk2bf(acc[6], acc[7]);
    *(uint4*)dst = v;
}

// ---------------------------------------------------------------------------
// Kernel 1: per-(h,n) setup: dt*lambda (for on-the-fly powers), W, r^64
// ---------------------------------------------------------------------------
__global__ __launch_bounds__(256) void setup_hn(
    const float* __restrict__ lre, const float* __restrict__ lim,
    const float* __restrict__ wre_in, const float* __restrict__ wim_in,
    const float* __restrict__ log_dt,
    float* __restrict__ dta, float* __restrict__ dtb,
    float* __restrict__ wre, float* __restrict__ wim,
    float* __restrict__ rcre, float* __restrict__ rcim)
{
    int idx = blockIdx.x * 256 + threadIdx.x;
    if (idx >= HN) return;
    int h = idx >> 5;
    float dt = expf(log_dt[h]);
    float ar = lre[idx], ai = lim[idx];
    float a = dt * ar, b = dt * ai;
    dta[idx] = a; dtb[idx] = b;
    // r^64
    float e64 = expf(64.0f * a);
    rcre[idx] = e64 * cosf(64.0f * b);
    rcim[idx] = e64 * sinf(64.0f * b);
    // W = (w)*(r-1)/lambda
    float er = expf(a);
    float rr = er * cosf(b), ri = er * sinf(b);
    float nr = rr - 1.0f, ni = ri;
    float den = ar * ar + ai * ai;
    float qr = (nr * ar + ni * ai) / den;
    float qi = (ni * ar - nr * ai) / den;
    float br = wre_in[idx], bi = wim_in[idx];
    wre[idx] = br * qr - bi * qi;
    wim[idx] = br * qi + bi * qr;
}

// ---------------------------------------------------------------------------
// Kernel 2: k_h[d] = 2*Re( sum_n W_n r_n^d ), d < 64
// ---------------------------------------------------------------------------
__global__ __launch_bounds__(256) void build_kh(
    const float* __restrict__ dta, const float* __restrict__ dtb,
    const float* __restrict__ wre, const float* __restrict__ wim,
    float* __restrict__ kh)
{
    int idx = blockIdx.x * 256 + threadIdx.x;   // (h,d)
    if (idx >= HDIM * CHUNK) return;
    int d = idx & 63, h = idx >> 6;
    float fd = (float)d;
    float acc = 0.0f;
    for (int n = 0; n < 32; ++n) {
        int hn = h * 32 + n;
        float er = expf(fd * dta[hn]);
        float c  = cosf(fd * dtb[hn]);
        float s  = sinf(fd * dtb[hn]);
        acc += wre[hn] * er * c - wim[hn] * er * s;
    }
    kh[idx] = 2.0f * acc;
}

// ---------------------------------------------------------------------------
// Kernel 3: P_h[m][i]  (m<32: Re r_m^{63-i} ; m>=32: Im r_{m-32}^{63-i})
// ---------------------------------------------------------------------------
__global__ __launch_bounds__(256) void build_P(
    const float* __restrict__ dta, const float* __restrict__ dtb,
    float* __restrict__ Pm)
{
    int idx = blockIdx.x * 256 + threadIdx.x;    // h*4096 + m*64 + i
    if (idx >= HDIM * 64 * 64) return;
    int i = idx & 63, m = (idx >> 6) & 63, h = idx >> 12;
    float e = (float)(63 - i);
    int hn = h * 32 + (m & 31);
    float er = expf(e * dta[hn]);
    float v = (m < 32) ? er * cosf(e * dtb[hn]) : er * sinf(e * dtb[hn]);
    Pm[idx] = v;
}

// ---------------------------------------------------------------------------
// Kernel 4: [T|Q]_h[i][kc] (64x128 per head)
//   kc<64 : Toeplitz  T[i][kc] = k_h[i-kc] for kc<=i else 0
//   kc>=64: m=kc-64;  m<32: 2Re(W_m r^{i+1}), m>=32: -2Im(W_{m-32} r^{i+1})
// ---------------------------------------------------------------------------
__global__ __launch_bounds__(256) void build_TQ(
    const float* __restrict__ dta, const float* __restrict__ dtb,
    const float* __restrict__ wre, const float* __restrict__ wim,
    const float* __restrict__ kh, float* __restrict__ TQm)
{
    int idx = blockIdx.x * 256 + threadIdx.x;    // h*8192 + i*128 + kc
    if (idx >= HDIM * 64 * 128) return;
    int kc = idx & 127, i = (idx >> 7) & 63, h = idx >> 13;
    float v;
    if (kc < 64) {
        v = (kc <= i) ? kh[h * 64 + (i - kc)] : 0.0f;
    } else {
        int m = kc - 64;
        int hn = h * 32 + (m & 31);
        float e = (float)(i + 1);
        float er = expf(e * dta[hn]);
        float c = cosf(e * dtb[hn]), s = sinf(e * dtb[hn]);
        if (m < 32) v =  2.0f * (wre[hn] * er * c - wim[hn] * er * s);
        else        v = -2.0f * (wre[hn] * er * s + wim[hn] * er * c);
    }
    TQm[idx] = v;
}

// ---------------------------------------------------------------------------
// Kernel 5 (generic): pack a row-major f32 matrix into the bf16 WMMA
// A-fragment layout with contiguous k order (k = 16*(lane/16) + e).
// ---------------------------------------------------------------------------
__global__ __launch_bounds__(256) void pack_frag(
    const float* __restrict__ src, unsigned short* __restrict__ dst,
    int ktiles, int ld, int total)
{
    int idx = blockIdx.x * 256 + threadIdx.x;
    if (idx >= total) return;
    int e    = idx & 15;
    int lane = (idx >> 4) & 31;
    int tile = idx >> 9;
    int kt   = tile % ktiles;
    int mt   = tile / ktiles;
    int v = e >> 1, hb = e & 1;
    int m = mt * 16 + (lane & 15);
    int k = kt * 32 + (lane >> 4) * 16 + (2 * v + hb);
    dst[idx] = f2bf(src[(size_t)m * ld + k]);
}

// ---------------------------------------------------------------------------
// Kernel 6: u (f32) -> ubf (bf16), 8 elements/thread
// ---------------------------------------------------------------------------
__global__ __launch_bounds__(256) void conv_u(
    const float* __restrict__ u, unsigned short* __restrict__ ubf)
{
    int idx = blockIdx.x * 256 + threadIdx.x;
    if (idx >= (BATCH * HDIM * LSEQ) / 8) return;
    const float4* up = (const float4*)u;
    float4 x = up[idx * 2], y = up[idx * 2 + 1];
    uint4 o;
    o.x = pk2bf(x.x, x.y);
    o.y = pk2bf(x.z, x.w);
    o.z = pk2bf(y.x, y.y);
    o.w = pk2bf(y.z, y.w);
    ((uint4*)ubf)[idx] = o;
}

// ---------------------------------------------------------------------------
// Kernel 7: chunk sums  S_h = P_h(64x64) @ U_h(64x1024).  No LDS: B-frag
// K-dim (position-in-chunk) is contiguous in ubf, so each lane loads its
// 16 bf16 as one 32B vector.  Columns c = b*64 + j; blockIdx.x = b.
// ---------------------------------------------------------------------------
__global__ __launch_bounds__(256) void gemm_chunksum(
    const unsigned short* __restrict__ packP,
    const unsigned short* __restrict__ ubf,
    unsigned short* __restrict__ Sc)
{
    int h = blockIdx.y, b = blockIdx.x;
    int tid = threadIdx.x, w = tid >> 5, lane = tid & 31;
    int mt = w & 3, nt0 = (w >> 2) * 2;
    int pl = lane & 15, hi = lane >> 4;
    int j0 = nt0 * 16 + pl, j1 = j0 + 16;
    const unsigned short* ub = ubf + ((size_t)(b * HDIM + h)) * LSEQ;
    v8f acc0 = {}; v8f acc1 = {};
    #pragma unroll
    for (int kb = 0; kb < 2; ++kb) {
        v16bf a  = *(const v16bf*)(packP + (size_t)h * 4096
                                   + ((mt * 2 + kb) * 32 + lane) * 16);
        v16bf b0 = *(const v16bf*)(ub + j0 * 64 + kb * 32 + hi * 16);
        v16bf b1 = *(const v16bf*)(ub + j1 * 64 + kb * 32 + hi * 16);
        acc0 = __builtin_amdgcn_wmma_f32_16x16x32_bf16(false, a, false, b0,
                (short)0, acc0, false, false);
        acc1 = __builtin_amdgcn_wmma_f32_16x16x32_bf16(false, a, false, b1,
                (short)0, acc1, false, false);
    }
    int rowb = mt * 16 + hi * 8;
    store8bf(Sc + ((size_t)(h * NCOL + b * 64 + j0)) * 64 + rowb, acc0);
    store8bf(Sc + ((size_t)(h * NCOL + b * 64 + j1)) * 64 + rowb, acc1);
}

// ---------------------------------------------------------------------------
// Kernel 8: chunk-level scan, IN PLACE.  wave = (h,b), lane = state n.
// Reads S_j once, then overwrites the same slot with carry_j (the state
// BEFORE chunk j), stored column-major so GEMM2 B-frags are contiguous.
// ---------------------------------------------------------------------------
__global__ __launch_bounds__(32) void chunk_combine(
    unsigned short* __restrict__ Sc,
    const float* __restrict__ rcre, const float* __restrict__ rcim)
{
    int hb = blockIdx.x;
    int h = hb >> 4, b = hb & 15;
    int lane = threadIdx.x;
    float Rr = rcre[h * 32 + lane], Ri = rcim[h * 32 + lane];
    float cr = 0.0f, ci = 0.0f;
    for (int j = 0; j < NCHUNK; ++j) {
        size_t col = ((size_t)(h * NCOL + b * 64 + j)) * 64;
        float sre = bf2f(Sc[col + lane]);
        float sim = bf2f(Sc[col + 32 + lane]);
        Sc[col + lane]      = f2bf(cr);
        Sc[col + 32 + lane] = f2bf(ci);
        float ncr = fmaf(Rr, cr, fmaf(-Ri, ci, sre));
        float nci = fmaf(Rr, ci, fmaf( Ri, cr, sim));
        cr = ncr; ci = nci;
    }
}

// ---------------------------------------------------------------------------
// Kernel 9: y_h = [T|Q]_h(64x128) @ [U ; Carry](128x1024); writes y as bf16
// into zy (channel-major for the final GEMM).
// ---------------------------------------------------------------------------
__global__ __launch_bounds__(256) void gemm_ylocal(
    const unsigned short* __restrict__ packTQ,
    const unsigned short* __restrict__ ubf,
    const unsigned short* __restrict__ Sc,
    unsigned short* __restrict__ zy)
{
    int h = blockIdx.y, b = blockIdx.x;
    int tid = threadIdx.x, w = tid >> 5, lane = tid & 31;
    int mt = w & 3, nt0 = (w >> 2) * 2;
    int pl = lane & 15, hi = lane >> 4;
    int j0 = nt0 * 16 + pl, j1 = j0 + 16;
    const unsigned short* ub = ubf + ((size_t)(b * HDIM + h)) * LSEQ;
    v8f acc0 = {}; v8f acc1 = {};
    #pragma unroll
    for (int kb = 0; kb < 4; ++kb) {
        v16bf a = *(const v16bf*)(packTQ + (size_t)h * 8192
                                  + ((mt * 4 + kb) * 32 + lane) * 16);
        v16bf b0, b1;
        if (kb < 2) {
            b0 = *(const v16bf*)(ub + j0 * 64 + kb * 32 + hi * 16);
            b1 = *(const v16bf*)(ub + j1 * 64 + kb * 32 + hi * 16);
        } else {
            b0 = *(const v16bf*)(Sc
                 + ((size_t)(h * NCOL + b * 64 + j0)) * 64 + (kb - 2) * 32 + hi * 16);
            b1 = *(const v16bf*)(Sc
                 + ((size_t)(h * NCOL + b * 64 + j1)) * 64 + (kb - 2) * 32 + hi * 16);
        }
        acc0 = __builtin_amdgcn_wmma_f32_16x16x32_bf16(false, a, false, b0,
                (short)0, acc0, false, false);
        acc1 = __builtin_amdgcn_wmma_f32_16x16x32_bf16(false, a, false, b1,
                (short)0, acc1, false, false);
    }
    int rowb = mt * 16 + hi * 8;    // position-in-chunk, 8 consecutive
    unsigned short* zb = zy + ((size_t)(b * HDIM + h)) * LSEQ;
    store8bf(zb + j0 * 64 + rowb, acc0);
    store8bf(zb + j1 * 64 + rowb, acc1);
}

// ---------------------------------------------------------------------------
// Kernel 10: final conv1x1 GEMM (c = conv_w @ [y;u] + b) fused with GLU.
// z channels k<256 read from zy, k>=256 from ubf; B strip staged through
// LDS in B-fragment layout, contiguous k (k = 16*(lane/16) + element); GLU pairing resolved in 32KB LDS.
// ---------------------------------------------------------------------------
__global__ __launch_bounds__(256) void gemm_glu(
    const unsigned short* __restrict__ packedA,
    const unsigned short* __restrict__ zy,
    const unsigned short* __restrict__ ubf,
    const float* __restrict__ conv_b,
    float* __restrict__ out)
{
    __shared__ unsigned short ldsB[4 * 32 * 16];
    __shared__ float ldsC[8 * 4 * 16 * 16];

    int nb = blockIdx.x;           // 0..1023 : 64-position strips
    int mb = blockIdx.y;           // 0..3    : paired GLU row blocks
    int tid = threadIdx.x;
    int w = tid >> 5, lane = tid & 31;
    int half = w >> 2, mt_local = w & 3;
    int mtA = half * 16 + mb * 4 + mt_local;
    int qb = nb * 64;
    int bq = qb >> 12;
    int lbase = qb & (LSEQ - 1);

    v8f acc0 = {}; v8f acc1 = {}; v8f acc2 = {}; v8f acc3 = {};

    int kk = tid >> 3;
    int c0 = (tid & 7) << 3;
    int hi = kk >> 4, rem = kk & 15;

    for (int kb = 0; kb < 16; ++kb) {
        int k = (kb << 5) + kk;
        const unsigned short* rowp = (k < HDIM)
            ? (zy  + ((size_t)(bq * HDIM + k)) * LSEQ)
            : (ubf + ((size_t)(bq * HDIM + (k - HDIM))) * LSEQ);
        union { uint4 v; unsigned short s[8]; } uu;
        uu.v = *(const uint4*)(rowp + lbase + c0);
        #pragma unroll
        for (int e = 0; e < 8; ++e) {
            int p = c0 + e;
            int nt = p >> 4, pl = p & 15;
            ldsB[(nt * 32 + hi * 16 + pl) * 16 + rem] = uu.s[e];
        }
        __syncthreads();

        v16bf afrag = *(const v16bf*)(packedA
            + ((size_t)(mtA * 16 + kb) * 32 + lane) * 16);
        if (kb < 15)
            __builtin_prefetch(packedA
                + ((size_t)(mtA * 16 + kb + 1) * 32 + lane) * 16, 0, 3);
        v16bf b0 = *(const v16bf*)(&ldsB[(0 * 32 + lane) * 16]);
        v16bf b1 = *(const v16bf*)(&ldsB[(1 * 32 + lane) * 16]);
        v16bf b2 = *(const v16bf*)(&ldsB[(2 * 32 + lane) * 16]);
        v16bf b3 = *(const v16bf*)(&ldsB[(3 * 32 + lane) * 16]);

        acc0 = __builtin_amdgcn_wmma_f32_16x16x32_bf16(false, afrag, false, b0,
                (short)0, acc0, false, false);
        acc1 = __builtin_amdgcn_wmma_f32_16x16x32_bf16(false, afrag, false, b1,
                (short)0, acc1, false, false);
        acc2 = __builtin_amdgcn_wmma_f32_16x16x32_bf16(false, afrag, false, b2,
                (short)0, acc2, false, false);
        acc3 = __builtin_amdgcn_wmma_f32_16x16x32_bf16(false, afrag, false, b3,
                (short)0, acc3, false, false);
        __syncthreads();
    }

    int mrow = mtA * 16;
    int rhalf = (lane >> 4) * 8;
    int cc = lane & 15;
    #pragma unroll
    for (int e = 0; e < 8; ++e) {
        int r = e + rhalf;
        float bias = conv_b[mrow + r];
        ldsC[((w * 4 + 0) * 16 + r) * 16 + cc] = acc0[e] + bias;
        ldsC[((w * 4 + 1) * 16 + r) * 16 + cc] = acc1[e] + bias;
        ldsC[((w * 4 + 2) * 16 + r) * 16 + cc] = acc2[e] + bias;
        ldsC[((w * 4 + 3) * 16 + r) * 16 + cc] = acc3[e] + bias;
    }
    __syncthreads();

    int col = tid & 63;
    int rb = (tid >> 6) << 4;
    int nt = col >> 4, cl = col & 15;
    #pragma unroll
    for (int i = 0; i < 16; ++i) {
        int r = rb + i;
        int wa = r >> 4, rr2 = r & 15;
        float av = ldsC[((wa * 4 + nt) * 16 + rr2) * 16 + cl];
        float gv = ldsC[(((wa + 4) * 4 + nt) * 16 + rr2) * 16 + cl];
        out[((size_t)(bq * HDIM + mb * 64 + r)) * LSEQ + lbase + col]
            = av / (1.0f + expf(-gv));
    }
}

// ---------------------------------------------------------------------------
extern "C" void kernel_launch(void* const* d_in, const int* in_sizes, int n_in,
                              void* d_out, int out_size, void* d_ws, size_t ws_size,
                              hipStream_t stream)
{
    const float* u         = (const float*)d_in[0];
    const float* lambda_re = (const float*)d_in[1];
    const float* lambda_im = (const float*)d_in[2];
    const float* w_re      = (const float*)d_in[3];
    const float* w_im      = (const float*)d_in[4];
    const float* log_dt    = (const float*)d_in[5];
    const float* conv_w    = (const float*)d_in[6];
    const float* conv_b    = (const float*)d_in[7];
    float* out = (float*)d_out;

    char* ws = (char*)d_ws;
    // small f32 tables
    float* dta  = (float*)(ws + 0);
    float* dtb  = (float*)(ws + 32768);
    float* wre  = (float*)(ws + 65536);
    float* wim  = (float*)(ws + 98304);
    float* rcre = (float*)(ws + 131072);
    float* rcim = (float*)(ws + 163840);
    float* kh   = (float*)(ws + 196608);                       // 64 KB
    unsigned short* packW  = (unsigned short*)(ws + 262144);   // 512 KB
    unsigned short* packP  = (unsigned short*)(ws + 786432);   // 2 MB
    unsigned short* packTQ = (unsigned short*)(ws + 2883584);  // 4 MB
    float* Pm  = (float*)(ws + 7340032);                       // 4 MB
    float* TQm = (float*)(ws + 11534336);                      // 8 MB
    unsigned short* ubf = (unsigned short*)(ws + 19922944);    // 32 MB
    unsigned short* Sc  = (unsigned short*)(ws + 53477376);    // 32 MB (S, then carry in place)
    unsigned short* zy  = (unsigned short*)(ws + 87031808);    // 32 MB

    setup_hn<<<HN / 256, 256, 0, stream>>>(lambda_re, lambda_im, w_re, w_im,
                                           log_dt, dta, dtb, wre, wim, rcre, rcim);
    build_kh<<<(HDIM * CHUNK) / 256, 256, 0, stream>>>(dta, dtb, wre, wim, kh);
    build_P<<<(HDIM * 64 * 64) / 256, 256, 0, stream>>>(dta, dtb, Pm);
    build_TQ<<<(HDIM * 64 * 128) / 256, 256, 0, stream>>>(dta, dtb, wre, wim, kh, TQm);

    pack_frag<<<(HDIM * 4096) / 256, 256, 0, stream>>>(Pm,  packP,  2,  64,  HDIM * 4096);
    pack_frag<<<(HDIM * 8192) / 256, 256, 0, stream>>>(TQm, packTQ, 4, 128,  HDIM * 8192);
    pack_frag<<<(C2 * C2)     / 256, 256, 0, stream>>>(conv_w, packW, 16, C2, C2 * C2);

    conv_u<<<(BATCH * HDIM * LSEQ / 8) / 256, 256, 0, stream>>>(u, ubf);

    gemm_chunksum<<<dim3(BATCH, HDIM), 256, 0, stream>>>(packP, ubf, Sc);
    chunk_combine<<<HDIM * BATCH, 32, 0, stream>>>(Sc, rcre, rcim);
    gemm_ylocal<<<dim3(BATCH, HDIM), 256, 0, stream>>>(packTQ, ubf, Sc, zy);

    gemm_glu<<<dim3((BATCH * LSEQ) / 64, HDIM / 64), 256, 0, stream>>>(
        packW, zy, ubf, conv_b, out);
}
